// DecoderRNN_90400471646369
// MI455X (gfx1250) — matrix operand order repaired and checked by reading, not compiled, not run on hardware
//
#include <hip/hip_runtime.h>
#include <hip/hip_bf16.h>
#include <math.h>

// ---------------------------------------------------------------------------
// DecoderRNN (show-attend-tell style) for MI455X / gfx1250, wave32 + WMMA bf16
// ---------------------------------------------------------------------------

typedef __attribute__((ext_vector_type(16))) __bf16 v16bf;
typedef __attribute__((ext_vector_type(8)))  __bf16 v8bf;
typedef __attribute__((ext_vector_type(8)))  float  v8f;

static constexpr int Bsz  = 64;
static constexpr int Tlen = 32;
static constexpr int Emb  = 512;
static constexpr int Hid  = 512;
static constexpr int Voc  = 32000;
static constexpr int Lreg = 49;

#define DEV_INLINE __device__ __forceinline__

DEV_INLINE float sigmoidf_(float x) { return 1.0f / (1.0f + __expf(-x)); }

DEV_INLINE v16bf join16(v8bf lo, v8bf hi) {
  v16bf r;
#pragma unroll
  for (int i = 0; i < 8; ++i) { r[i] = lo[i]; r[8 + i] = hi[i]; }
  return r;
}

// ---- WMMA fragment loads (contiguous-K order k = 16h + s, not ISA table order; wave32) ----------------
// A: 16x32 bf16 (MxK). lanes 0-15: row=lane, v0..7=K{0..15};
//                      lanes 16-31: row=lane-16, K{16..31}.
DEV_INLINE v16bf load_a_frag(const __bf16* __restrict__ A, int lda,
                             int m0, int k, int lane) {
  int row = m0 + (lane & 15);
  int kk  = (lane >> 4) * 16;         // 0 or 16
  const __bf16* p = A + (size_t)row * lda + k + kk;
  return join16(*(const v8bf*)(p), *(const v8bf*)(p + 8));
}

// B: 32x16 bf16 (KxN), sourced from row-major W[N,K] (i.e. B = W^T).
// lanes 0-15: col=lane, v0..7 = K{0..15}; lanes 16-31: col=lane-16, K{16..31}.
DEV_INLINE v16bf load_b_frag(const __bf16* __restrict__ W, int ldw,
                             int n0, int k, int lane) {
  int n  = n0 + (lane & 15);
  int ko = (lane >> 4) * 16;          // 0 or 16
  const __bf16* p = W + (size_t)n * ldw + k + ko;
  return join16(*(const v8bf*)(p), *(const v8bf*)(p + 8));
}

DEV_INLINE v8f wmma_bf16(v16bf a, v16bf b, v8f c) {
  return __builtin_amdgcn_wmma_f32_16x16x32_bf16(false, a, false, b,
                                                 (short)0, c, false, false);
}

// ---- Generic wave-level bf16 GEMM, rolled-loop software pipeline -----------
// C[m,n] = sum_k A[m,k]*W[n,k] + bias[n].  EPI: 0 = f32, 1 = tanh->bf16.
// K is a RUNTIME parameter on purpose: it keeps the k-loop rolled, so the
// back-edge forces next-iteration loads (a_nxt/b_nxt) to issue before the
// current WMMAs -> full-iteration load lookahead with staggered partial
// s_wait_loadcnt (verified codegen); the rotation copies are dual-issue noise.
template <int NT, int EPI>
__global__ void __launch_bounds__(256)
gemm_bf16_wmma(const __bf16* __restrict__ A, int lda,
               const __bf16* __restrict__ W, int ldw,
               const float* __restrict__ bias,
               float* __restrict__ Cf, __bf16* __restrict__ Cb, int ldc,
               int M, int N, int K) {
  int lane = threadIdx.x & 31;
  int wave = blockIdx.x * (blockDim.x >> 5) + (threadIdx.x >> 5);
  int nGroups = N / (16 * NT);
  int mTiles  = M / 16;
  if (wave >= nGroups * mTiles) return;
  int m0 = (wave / nGroups) * 16;
  int n0 = (wave % nGroups) * (16 * NT);

  v8f acc[NT] = {};
  v16bf a_cur = load_a_frag(A, lda, m0, 0, lane);
  v16bf b_cur[NT];
#pragma unroll
  for (int j = 0; j < NT; ++j) b_cur[j] = load_b_frag(W, ldw, n0 + 16 * j, 0, lane);

  for (int k = 32; k < K; k += 32) {
    // next-iteration loads issue before current WMMAs consume cur fragments
    v16bf a_nxt = load_a_frag(A, lda, m0, k, lane);
    v16bf b_nxt[NT];
#pragma unroll
    for (int j = 0; j < NT; ++j) b_nxt[j] = load_b_frag(W, ldw, n0 + 16 * j, k, lane);
#pragma unroll
    for (int j = 0; j < NT; ++j) acc[j] = wmma_bf16(a_cur, b_cur[j], acc[j]);
    a_cur = a_nxt;
#pragma unroll
    for (int j = 0; j < NT; ++j) b_cur[j] = b_nxt[j];
  }
#pragma unroll
  for (int j = 0; j < NT; ++j) acc[j] = wmma_bf16(a_cur, b_cur[j], acc[j]);

  // D layout: lane 0-15 -> col=lane, rows m0+r; lane 16-31 -> col, rows m0+8+r.
  int nn = lane & 15;
  int mo = (lane >> 4) * 8;
#pragma unroll
  for (int j = 0; j < NT; ++j) {
    int n = n0 + 16 * j + nn;
    float bv = bias ? bias[n] : 0.0f;
#pragma unroll
    for (int r = 0; r < 8; ++r) {
      int gm = m0 + mo + r;
      float v = acc[j][r] + bv;
      if (EPI == 0) Cf[(size_t)gm * ldc + n] = v;
      else          Cb[(size_t)gm * ldc + n] = (__bf16)tanhf(v);
    }
  }
}

// ---- Fused LSTM: gates GEMM + cell nonlinearity in registers ---------------
// Wave owns m-tile (16 batch rows) x 16 hidden cols; its 4 accumulators are
// the gate-aligned n-tiles {j, j+512, j+1024, j+1536} (PyTorch i,f,g,o).
// A = [x | h_prev] bf16 [64,1024]; Wc = [W_ih | W_hh] bf16 [2048,1024].
// K passed at runtime (=1024) to keep the pipeline loop rolled (see above).
__global__ void __launch_bounds__(256)
lstm_fused(const __bf16* __restrict__ A, const __bf16* __restrict__ Wc,
           const float* __restrict__ bias, float* __restrict__ c,
           float* __restrict__ hf, __bf16* __restrict__ hb0,
           __bf16* __restrict__ hb1, int K) {
  int lane = threadIdx.x & 31;
  int wave = blockIdx.x * 8 + (threadIdx.x >> 5);
  int m0 = (wave >> 5) * 16;            // 4 m-tiles x 32 j-tiles = 128 waves
  int j0 = (wave & 31) * 16;

  v8f acc[4] = {};
  v16bf a_cur = load_a_frag(A, 1024, m0, 0, lane);
  v16bf b_cur[4];
#pragma unroll
  for (int g = 0; g < 4; ++g) b_cur[g] = load_b_frag(Wc, 1024, g * 512 + j0, 0, lane);

  for (int k = 32; k < K; k += 32) {
    v16bf a_nxt = load_a_frag(A, 1024, m0, k, lane);
    v16bf b_nxt[4];
#pragma unroll
    for (int g = 0; g < 4; ++g) b_nxt[g] = load_b_frag(Wc, 1024, g * 512 + j0, k, lane);
#pragma unroll
    for (int g = 0; g < 4; ++g) acc[g] = wmma_bf16(a_cur, b_cur[g], acc[g]);
    a_cur = a_nxt;
#pragma unroll
    for (int g = 0; g < 4; ++g) b_cur[g] = b_nxt[g];
  }
#pragma unroll
  for (int g = 0; g < 4; ++g) acc[g] = wmma_bf16(a_cur, b_cur[g], acc[g]);

  int nn = lane & 15;
  int mo = (lane >> 4) * 8;
  int j  = j0 + nn;
  float bi = bias[j],        bf_ = bias[512 + j];
  float bg = bias[1024 + j], bo  = bias[1536 + j];
#pragma unroll
  for (int r = 0; r < 8; ++r) {
    int m = m0 + mo + r;
    float ig = sigmoidf_(acc[0][r] + bi);
    float fg = sigmoidf_(acc[1][r] + bf_);
    float gg = tanhf(acc[2][r] + bg);
    float og = sigmoidf_(acc[3][r] + bo);
    size_t ci = (size_t)m * Hid + j;
    float cn = fg * c[ci] + ig * gg;
    c[ci] = cn;
    float h = og * tanhf(cn);
    if (hf)  hf[ci] = h;
    __bf16 hb = (__bf16)h;
    if (hb0) hb0[(size_t)m * 1024 + j] = hb;
    if (hb1) hb1[(size_t)m * 1024 + j] = hb;
  }
}

// ---- Vocab projection: block-tiled, LDS double-buffered --------------------
// C[2048,32000] = A[2048,512] @ W[32000,512]^T + b_out, remapped to [B,T,V].
// Block = 8 waves: BM=128 (wave m-tiles), BN=128, BK=32. W tile staged via
// LDS ping-pong (rows padded to 40 elems -> conflict-free stride-20 banks).
static constexpr int LDB = 40;          // padded bf16 row length for 32-K tile
__global__ void __launch_bounds__(256)
gemm_vocab(const __bf16* __restrict__ A, const __bf16* __restrict__ W,
           const float* __restrict__ bias, float* __restrict__ out) {
  __shared__ __bf16 lds[2][128 * LDB];  // 2 x 10 KB
  int tid  = threadIdx.x;
  int lane = tid & 31, wave = tid >> 5;
  int nBlk = Voc / 128;                 // 250
  int m0 = (blockIdx.x / nBlk) * 128 + wave * 16;
  int n0 = (blockIdx.x % nBlk) * 128;

  // staging: thread pair covers one W row chunk of 32 K-values
  int nloc = tid >> 1;
  int half = (tid & 1) * 16;
  const __bf16* wsrc = W + (size_t)(n0 + nloc) * Hid + half;

  v8f acc[8] = {};
  {
    v8bf r0 = *(const v8bf*)(wsrc);
    v8bf r1 = *(const v8bf*)(wsrc + 8);
    __bf16* d = &lds[0][0] + nloc * LDB + half;
    *(v8bf*)(d) = r0; *(v8bf*)(d + 8) = r1;
  }
  __syncthreads();

  constexpr int NK = Hid / 32;          // 16
#pragma unroll 2
  for (int ks = 0; ks < NK; ++ks) {
    const int cur = ks & 1;
    const bool more = (ks + 1) < NK;
    v8bf nr0{}, nr1{};
    if (more) {                         // global prefetch of next W tile
      const __bf16* p = wsrc + (ks + 1) * 32;
      nr0 = *(const v8bf*)(p); nr1 = *(const v8bf*)(p + 8);
    }
    v16bf a = load_a_frag(A, Hid, m0, ks * 32, lane);
    const __bf16* base = &lds[cur][0];
#pragma unroll
    for (int j = 0; j < 8; ++j) {
      const __bf16* bp = base + (16 * j + (lane & 15)) * LDB + (lane >> 4) * 16;
      v16bf b = join16(*(const v8bf*)(bp), *(const v8bf*)(bp + 8));
      acc[j] = wmma_bf16(a, b, acc[j]);
    }
    if (more) {
      __bf16* d = &lds[cur ^ 1][0] + nloc * LDB + half;
      *(v8bf*)(d) = nr0; *(v8bf*)(d + 8) = nr1;
    }
    __syncthreads();
  }

  int nn = lane & 15;
  int mo = (lane >> 4) * 8;
#pragma unroll
  for (int j = 0; j < 8; ++j) {
    int n = n0 + 16 * j + nn;
    float bv = bias[n];
#pragma unroll
    for (int r = 0; r < 8; ++r) {
      int gm = m0 + mo + r;             // gm = t*64 + b
      int bb = gm & 63, tt = gm >> 6;
      out[((size_t)bb * Tlen + tt) * Voc + n] = acc[j][r] + bv;
    }
  }
}

// ---- Attention: scores -> softmax -> context (one block per batch row) -----
__global__ void __launch_bounds__(256)
attention(const float* __restrict__ features, const float* __restrict__ h2,
          __bf16* __restrict__ ctx_dst) {
  __shared__ float sc[64];
  __shared__ float hsh[Hid];
  int b = blockIdx.x;
  for (int i = threadIdx.x; i < Hid; i += 256) hsh[i] = h2[(size_t)b * Hid + i];
  __syncthreads();

  int wave = threadIdx.x >> 5, lane = threadIdx.x & 31;
  for (int l = wave; l < Lreg; l += 8) {
    const float* fr = features + ((size_t)b * Lreg + l) * Hid;
    float s = 0.0f;
    for (int i = lane; i < Hid; i += 32) s += fr[i] * hsh[i];
#pragma unroll
    for (int off = 16; off > 0; off >>= 1) s += __shfl_down(s, off, 32);
    if (lane == 0) sc[l] = s;
  }
  __syncthreads();

  if (threadIdx.x == 0) {
    float mx = sc[0];
    for (int l = 1; l < Lreg; ++l) mx = fmaxf(mx, sc[l]);
    float s = 0.0f;
    for (int l = 0; l < Lreg; ++l) { float e = __expf(sc[l] - mx); sc[l] = e; s += e; }
    float inv = 1.0f / s;
    for (int l = 0; l < Lreg; ++l) sc[l] *= inv;
  }
  __syncthreads();

  for (int i = threadIdx.x; i < Hid; i += 256) {
    float a = 0.0f;
    for (int l = 0; l < Lreg; ++l)
      a += sc[l] * features[((size_t)b * Lreg + l) * Hid + i];
    ctx_dst[(size_t)b * 1024 + i] = (__bf16)a;
  }
}

// ---- In-place log-softmax over rows of length Voc --------------------------
__global__ void __launch_bounds__(256)
log_softmax_rows(float* __restrict__ out) {
  __shared__ float red[256];
  size_t base = (size_t)blockIdx.x * Voc;
  int tid = threadIdx.x;
  float m = -INFINITY;
  for (int i = tid; i < Voc; i += 256) m = fmaxf(m, out[base + i]);
  red[tid] = m; __syncthreads();
  for (int s = 128; s > 0; s >>= 1) {
    if (tid < s) red[tid] = fmaxf(red[tid], red[tid + s]);
    __syncthreads();
  }
  m = red[0]; __syncthreads();
  float sum = 0.0f;
  for (int i = tid; i < Voc; i += 256) sum += __expf(out[base + i] - m);
  red[tid] = sum; __syncthreads();
  for (int s = 128; s > 0; s >>= 1) {
    if (tid < s) red[tid] += red[tid + s];
    __syncthreads();
  }
  float lse = m + __logf(red[0]);
  for (int i = tid; i < Voc; i += 256) out[base + i] -= lse;
}

// ---- Conversion / setup kernels -------------------------------------------
__global__ void __launch_bounds__(256)
cvt_bf16(const float* __restrict__ src, __bf16* __restrict__ dst, size_t n) {
  size_t i = (size_t)blockIdx.x * blockDim.x + threadIdx.x;
  if (i < n) dst[i] = (__bf16)src[i];
}

__global__ void __launch_bounds__(256)
cvt_pair_bf16(const float* __restrict__ Wa, const float* __restrict__ Wb,
              __bf16* __restrict__ dst, int rows, int Ka, int Kb) {
  size_t i = (size_t)blockIdx.x * blockDim.x + threadIdx.x;
  int kt = Ka + Kb;
  if (i >= (size_t)rows * kt) return;
  int n = (int)(i / kt), k = (int)(i % kt);
  float v = (k < Ka) ? Wa[(size_t)n * Ka + k] : Wb[(size_t)n * Kb + (k - Ka)];
  dst[i] = (__bf16)v;
}

__global__ void __launch_bounds__(256)
add_bias_f32(const float* __restrict__ a, const float* __restrict__ b,
             float* __restrict__ d, int n) {
  int i = blockIdx.x * blockDim.x + threadIdx.x;
  if (i < n) d[i] = a[i] + b[i];
}

__global__ void __launch_bounds__(256)
embed_gather(const int* __restrict__ captions, const float* __restrict__ table,
             __bf16* __restrict__ xall1) {
  size_t i = (size_t)blockIdx.x * blockDim.x + threadIdx.x;
  if (i >= (size_t)Tlen * Bsz * Emb) return;
  int e = (int)(i % Emb);
  int b = (int)((i / Emb) % Bsz);
  int t = (int)(i / ((size_t)Emb * Bsz));
  int tok = captions[b * Tlen + t];
  xall1[((size_t)t * Bsz + b) * 1024 + e] = (__bf16)table[(size_t)tok * Emb + e];
}

// ---------------------------------------------------------------------------
extern "C" void kernel_launch(void* const* d_in, const int* in_sizes, int n_in,
                              void* d_out, int out_size, void* d_ws, size_t ws_size,
                              hipStream_t stream) {
  const float* features = (const float*)d_in[0];
  const int*   captions = (const int*)d_in[1];
  const float* table    = (const float*)d_in[2];
  const float* W_ih1 = (const float*)d_in[3];
  const float* W_hh1 = (const float*)d_in[4];
  const float* b_ih1 = (const float*)d_in[5];
  const float* b_hh1 = (const float*)d_in[6];
  const float* W_ih2 = (const float*)d_in[7];
  const float* W_hh2 = (const float*)d_in[8];
  const float* b_ih2 = (const float*)d_in[9];
  const float* b_hh2 = (const float*)d_in[10];
  const float* W_cat = (const float*)d_in[11];
  const float* b_cat = (const float*)d_in[12];
  const float* W_out = (const float*)d_in[13];
  const float* b_out = (const float*)d_in[14];
  float* out = (float*)d_out;

  // ---- workspace carve-up (~57 MB) ----
  char* w = (char*)d_ws;
  auto alloc = [&](size_t bytes) {
    char* p = w; w += (bytes + 255) & ~(size_t)255; return p;
  };
  __bf16* Wc1   = (__bf16*)alloc((size_t)2048 * 1024 * 2);  // [W_ih1|W_hh1]
  __bf16* Wc2   = (__bf16*)alloc((size_t)2048 * 1024 * 2);
  __bf16* Wcatb = (__bf16*)alloc((size_t)512  * 1024 * 2);
  __bf16* Woutb = (__bf16*)alloc((size_t)Voc  * Hid  * 2);
  float*  bc1   = (float*) alloc(2048 * 4);
  float*  bc2   = (float*) alloc(2048 * 4);
  size_t xbytes = (size_t)Tlen * Bsz * 1024 * 2;
  __bf16* xall1 = (__bf16*)alloc(xbytes);   // [t][b][ emb | h1_prev ]
  __bf16* xall2 = (__bf16*)alloc(xbytes);   // [t][b][ h1  | h2_prev ]
  __bf16* acat  = (__bf16*)alloc(xbytes);   // [t][b][ ctx | h2      ]
  __bf16* catAll= (__bf16*)alloc((size_t)Tlen * Bsz * Hid * 2);
  float*  c1    = (float*) alloc((size_t)Bsz * Hid * 4);
  float*  c2    = (float*) alloc((size_t)Bsz * Hid * 4);
  float*  h2f   = (float*) alloc((size_t)Bsz * Hid * 4);

  hipMemsetAsync(xall1, 0, xbytes, stream);
  hipMemsetAsync(xall2, 0, xbytes, stream);
  hipMemsetAsync(acat,  0, xbytes, stream);
  hipMemsetAsync(c1, 0, (size_t)Bsz * Hid * 4, stream);
  hipMemsetAsync(c2, 0, (size_t)Bsz * Hid * 4, stream);

  {
    size_t n = (size_t)2048 * 1024;
    cvt_pair_bf16<<<(int)((n + 255) / 256), 256, 0, stream>>>(W_ih1, W_hh1, Wc1, 2048, Emb, Hid);
    cvt_pair_bf16<<<(int)((n + 255) / 256), 256, 0, stream>>>(W_ih2, W_hh2, Wc2, 2048, Hid, Hid);
    size_t nc = (size_t)512 * 1024;
    cvt_bf16<<<(int)((nc + 255) / 256), 256, 0, stream>>>(W_cat, Wcatb, nc);
    size_t no = (size_t)Voc * Hid;
    cvt_bf16<<<(int)((no + 255) / 256), 256, 0, stream>>>(W_out, Woutb, no);
    add_bias_f32<<<8, 256, 0, stream>>>(b_ih1, b_hh1, bc1, 2048);
    add_bias_f32<<<8, 256, 0, stream>>>(b_ih2, b_hh2, bc2, 2048);
  }
  {
    size_t n = (size_t)Tlen * Bsz * Emb;
    embed_gather<<<(int)((n + 255) / 256), 256, 0, stream>>>(captions, table, xall1);
  }

  // ---- recurrence over T (4 launches / step) ----
  for (int t = 0; t < Tlen; ++t) {
    __bf16* x1 = xall1 + (size_t)t * Bsz * 1024;
    __bf16* x2 = xall2 + (size_t)t * Bsz * 1024;
    __bf16* ac = acat  + (size_t)t * Bsz * 1024;

    // LSTM1: fused gates GEMM + cell; h1 -> xall2[t][:,0:512] and next x-slot
    lstm_fused<<<16, 256, 0, stream>>>(
        x1, Wc1, bc1, c1, nullptr, x2,
        (t + 1 < Tlen) ? (xall1 + (size_t)(t + 1) * Bsz * 1024 + 512) : nullptr,
        1024);

    // LSTM2: h2 -> acat[t][:,512:1024] (+ next slot), f32 for attention
    lstm_fused<<<16, 256, 0, stream>>>(
        x2, Wc2, bc2, c2, h2f, ac + 512,
        (t + 1 < Tlen) ? (xall2 + (size_t)(t + 1) * Bsz * 1024 + 512) : nullptr,
        1024);

    // attention: ctx -> acat[t][:,0:512]
    attention<<<Bsz, 256, 0, stream>>>(features, h2f, ac);

    // concat = tanh([ctx|h2] @ W_cat^T + b_cat) -> bf16 rows of catAll
    gemm_bf16_wmma<4, 1><<<4, 256, 0, stream>>>(
        ac, 1024, Wcatb, 1024, b_cat, nullptr,
        catAll + (size_t)t * Bsz * Hid, Hid, Bsz, Hid, 1024);
  }

  // ---- vocab projection: (2048/128)*(32000/128) = 4000 blocks ----
  gemm_vocab<<<4000, 256, 0, stream>>>(catAll, Woutb, b_out, out);

  // ---- in-place log-softmax over every (b,t) row ----
  log_softmax_rows<<<Bsz * Tlen, 256, 0, stream>>>(out);
}
